// MultiDimensionalRNNBase_77635828842557
// MI455X (gfx1250) — matrix operand order repaired and checked by reading, not compiled, not run on hardware
//
#include <hip/hip_runtime.h>
#include <cmath>

typedef __bf16 bf16_t;
typedef bf16_t v16bf __attribute__((ext_vector_type(16)));
typedef float  v8f   __attribute__((ext_vector_type(8)));
typedef float  v4f   __attribute__((ext_vector_type(4)));
typedef int    v4i   __attribute__((ext_vector_type(4)));

#define HH 32
#define WW 32
#define HID 64
#define NCLS 10
#define BATCHN 128
#define STEPS 63            // W + H - 1

#if __has_builtin(__builtin_amdgcn_tanhf)
#define TANHF(x) __builtin_amdgcn_tanhf(x)
#else
#define TANHF(x) tanhf(x)
#endif

// ---- LDS layout ----
#define BT_STRIDE 136       // weight row stride, elements (128 + 8 pad) -> 272B, 16B aligned
#define H_STRIDE  72        // h row stride, elements (64 + 8 pad) -> 144B, 16B aligned
#define H_ROWS    33        // 1 zero guard row + 32
#define ACT_STRIDE 72       // activations [pix][o] stride
#define ACT_TRASH_PIX 1024  // extra row absorbing out-of-band stores
#define XSK_STRIDE 36       // skewed-x [c][r] row stride (floats), 144B -> 16B aligned

#define BT_BYTES   (HID * BT_STRIDE * 2)            // 17408
#define XSK_BYTES  (64 * XSK_STRIDE * 4)            // 9216
#define HBUF_ELEMS (H_ROWS * H_STRIDE)              // 2376
#define HB_BYTES   (2 * HBUF_ELEMS * 2)             // 9504 (ping-pong)
#define ACT_BYTES  ((HH * WW + 1) * ACT_STRIDE * 2) // 147600 (incl. trash row)
#define RED_BYTES  (8 * NCLS * 4)                   // 320
#define SMEM_BYTES (BT_BYTES + XSK_BYTES + HB_BYTES + ACT_BYTES + RED_BYTES) // 184048

union Frag { v16bf v; v4i q[2]; };

__global__ __launch_bounds__(256) void mdrnn_kernel(
    const float* __restrict__ x,
    const float* __restrict__ w_in,
    const float* __restrict__ b_in,
    const float* __restrict__ w_state,
    const float* __restrict__ b_state,
    const float* __restrict__ w_fc,
    float* __restrict__ partial)   // [512][10] partial logits
{
  extern __shared__ char smem[];
  bf16_t* Bt    = (bf16_t*)smem;                                         // [64][136]  Bt[o][k]
  float*  xsk   = (float*)(smem + BT_BYTES);                             // [64][36]   xsk[c][r]
  bf16_t* hbuf  = (bf16_t*)(smem + BT_BYTES + XSK_BYTES);                // 2 x [33][72] ping-pong h
  bf16_t* actsL = (bf16_t*)(smem + BT_BYTES + XSK_BYTES + HB_BYTES);     // [1025][72] activations
  float*  redL  = (float*)(smem + BT_BYTES + XSK_BYTES + HB_BYTES + ACT_BYTES); // [8][10]

  const int img  = blockIdx.x;          // dir*128 + b
  const int dir  = img >> 7;
  const int bb   = img & 127;
  const int tid  = threadIdx.x;
  const int lane = tid & 31;
  const int wid  = tid >> 5;
  const int mt   = wid & 1;             // row tile (0..1)
  const int nt   = wid >> 1;            // out-channel tile (0..3)

  // ---- stage concatenated weights: Bt[o][k] = k<64 ? w_up[o][k] : w_same[o][k-64]
  for (int i = tid; i < HID * 128; i += 256) {
    int o = i >> 7, k = i & 127;
    float v = w_state[((size_t)(o * HID + (k & 63))) * 2 + (k >> 6)];
    Bt[o * BT_STRIDE + k] = (bf16_t)v;
  }
  // ---- stage skewed x, transposed [c][r]: xsk[c][r] = x[fr(r)][fc(c-r)] if 0<=c-r<32 else 0
  {
    const bool flr = (dir == 1) || (dir == 3);
    const bool flc = (dir == 2) || (dir == 3);
    const float* xb = x + (size_t)bb * (HH * WW);
    for (int i = tid; i < 64 * HH; i += 256) {
      int c = i >> 5, r = i & 31;
      int j = c - r;
      float v = 0.0f;
      if (j >= 0 && j < WW) {
        int rr = flr ? (HH - 1 - r) : r;
        int jj = flc ? (WW - 1 - j) : j;
        v = xb[rr * WW + jj];
      }
      xsk[c * XSK_STRIDE + r] = v;
    }
  }
  // ---- zero both h buffers (guard row 0 of each stays zero forever)
  for (int i = tid; i < 2 * HBUF_ELEMS; i += 256) hbuf[i] = (bf16_t)0.0f;

  // ---- per-lane constants (WMMA layouts)
  const int lm     = lane & 15;
  const int hi     = lane >> 4;             // 0 | 1
  const int o_lane = nt * 16 + lm;          // output channel (B/C/D column)
  const float win_o  = w_in[o_lane];
  const float bias_o = b_in[o_lane] + b_state[o_lane];
  const int kb    = hi * 8;                 // A-frag K sub-base
  const int r_a   = mt * 16 + lm;           // A-frag row (M)
  const int rbase = mt * 16 + hi * 8;       // C/D row base for acc[v]

  __syncthreads();

  // ---- preload B fragments: lane holds N=o_lane, K = [hi*8, hi*8+8) and [16+hi*8, 16+hi*8+8)
  Frag bfrag[4];
  {
    const char* brow = (const char*)(Bt + o_lane * BT_STRIDE);
#pragma unroll
    for (int kt = 0; kt < 4; ++kt) {
      const char* p = brow + (kt * 32 + hi * 8) * 2;
      bfrag[kt].q[0] = *(const v4i*)(p);
      bfrag[kt].q[1] = *(const v4i*)(p + 32);
    }
  }

  // ---- 63-step wavefront recurrence
  for (int c = 0; c < STEPS; ++c) {
    const bf16_t* hin  = hbuf + ((c & 1) ? HBUF_ELEMS : 0);
    bf16_t*       hout = hbuf + ((c & 1) ? 0 : HBUF_ELEMS);

    // load all A fragments up front (K tiles 0,1 = h_up via guard row; 2,3 = h_same)
    Frag af[4];
#pragma unroll
    for (int kt = 0; kt < 4; ++kt) {
      int hrow = (kt < 2) ? r_a : (r_a + 1);
      const char* p = (const char*)(hin + hrow * H_STRIDE) + ((kt & 1) * 32 + kb) * 2;
      af[kt].q[0] = *(const v4i*)(p);        // K = kb .. kb+7
      af[kt].q[1] = *(const v4i*)(p + 32);   // K = kb+16 .. kb+23
    }
    // x gather for this wave: 8 contiguous floats at xsk[c][rbase..rbase+7]
    const float* xcol = xsk + c * XSK_STRIDE + rbase;
    v4f xv0 = *(const v4f*)(xcol);
    v4f xv1 = *(const v4f*)(xcol + 4);

    // two independent accumulation chains on the XDL pipe
    v8f acc0 = {}, acc1 = {};
    acc0 = __builtin_amdgcn_wmma_f32_16x16x32_bf16(false, af[0].v, false, bfrag[0].v, (short)0, acc0, false, false);
    acc1 = __builtin_amdgcn_wmma_f32_16x16x32_bf16(false, af[1].v, false, bfrag[1].v, (short)0, acc1, false, false);
    acc0 = __builtin_amdgcn_wmma_f32_16x16x32_bf16(false, af[2].v, false, bfrag[2].v, (short)0, acc0, false, false);
    acc1 = __builtin_amdgcn_wmma_f32_16x16x32_bf16(false, af[3].v, false, bfrag[3].v, (short)0, acc1, false, false);

    // phase 1: all 8 tanh back-to-back (TRANS pipeline)
    float hn[8];
#pragma unroll
    for (int v = 0; v < 8; ++v) {
      float xval = (v < 4) ? xv0[v] : xv1[v - 4];
      float s = fmaf(win_o, xval, acc0[v] + acc1[v] + bias_o);
      hn[v] = TANHF(s);
    }
    // phase 2: branch-free stores (invalid lanes land in the trash row)
#pragma unroll
    for (int v = 0; v < 8; ++v) {
      int rr = rbase + v;
      int j  = c - rr;
      bf16_t hb = (bf16_t)hn[v];
      hout[(rr + 1) * H_STRIDE + o_lane] = hb;
      int apix = ((unsigned)j < (unsigned)WW) ? (rr * WW + j) : ACT_TRASH_PIX;
      actsL[apix * ACT_STRIDE + o_lane] = hb;
    }
    __syncthreads();
  }

  // ---- fused FC: partial[img][cls] = sum_f acts[f] * w_fc[dir*65536 + f][cls]
  float pacc[NCLS];
#pragma unroll
  for (int k = 0; k < NCLS; ++k) pacc[k] = 0.0f;

  const size_t dirOff = (size_t)dir * (HID * HH * WW);
  for (int idx = tid; idx < HID * HH * WW; idx += 256) {
    int o = idx >> 10, pix = idx & 1023;                 // f = o*1024 + pix
    float av = (float)actsL[pix * ACT_STRIDE + o];
    const float* wrow = w_fc + (dirOff + (size_t)idx) * NCLS;
    __builtin_prefetch((const void*)(wrow + 256 * NCLS), 0, 1);
    const float2* w2 = (const float2*)wrow;              // 8B aligned (f*40 bytes)
#pragma unroll
    for (int q = 0; q < 5; ++q) {
      float2 wv = w2[q];
      pacc[2 * q]     += av * wv.x;
      pacc[2 * q + 1] += av * wv.y;
    }
  }

  // wave32 shuffle reduce, then cross-wave via LDS
#pragma unroll
  for (int k = 0; k < NCLS; ++k) {
    float v = pacc[k];
    for (int off = 16; off > 0; off >>= 1) v += __shfl_xor(v, off, 32);
    pacc[k] = v;
  }
  if (lane == 0) {
#pragma unroll
    for (int k = 0; k < NCLS; ++k) redL[wid * NCLS + k] = pacc[k];
  }
  __syncthreads();
  if (tid < NCLS) {
    float s = 0.0f;
#pragma unroll
    for (int w = 0; w < 8; ++w) s += redL[w * NCLS + tid];
    partial[(size_t)img * NCLS + tid] = s;
  }
}

__global__ __launch_bounds__(32) void head_kernel(
    const float* __restrict__ partial,
    const float* __restrict__ b_fc,
    float* __restrict__ out)
{
  const int b = blockIdx.x;
  const int t = threadIdx.x;
  float v = -__builtin_inff();
  if (t < NCLS) {
    float s = b_fc[t];
#pragma unroll
    for (int d = 0; d < 4; ++d)
      s += partial[((size_t)d * BATCHN + b) * NCLS + t];
    v = s;
  }
  float m = v;
  for (int off = 16; off > 0; off >>= 1) m = fmaxf(m, __shfl_xor(m, off, 32));
  float e = (t < NCLS) ? __expf(v - m) : 0.0f;
  float z = e;
  for (int off = 16; off > 0; off >>= 1) z += __shfl_xor(z, off, 32);
  if (t < NCLS) out[b * NCLS + t] = (v - m) - __logf(z);
}

extern "C" void kernel_launch(void* const* d_in, const int* in_sizes, int n_in,
                              void* d_out, int out_size, void* d_ws, size_t ws_size,
                              hipStream_t stream) {
  const float* x       = (const float*)d_in[0];
  const float* w_in    = (const float*)d_in[1];
  const float* b_in    = (const float*)d_in[2];
  const float* w_state = (const float*)d_in[3];
  const float* b_state = (const float*)d_in[4];
  const float* w_fc    = (const float*)d_in[5];
  const float* b_fc    = (const float*)d_in[6];
  float* out     = (float*)d_out;
  float* partial = (float*)d_ws;            // 512*10 floats = 20480 B

  (void)in_sizes; (void)n_in; (void)out_size; (void)ws_size;

  // Opt in to large dynamic LDS (CDNA5 WGP supports up to 320 KB).
  hipFuncSetAttribute((const void*)mdrnn_kernel,
                      hipFuncAttributeMaxDynamicSharedMemorySize, SMEM_BYTES);

  mdrnn_kernel<<<4 * BATCHN, 256, SMEM_BYTES, stream>>>(
      x, w_in, b_in, w_state, b_state, w_fc, partial);
  head_kernel<<<BATCHN, 32, 0, stream>>>(partial, b_fc, out);
}
